// ACT_basic_15092515078271
// MI455X (gfx1250) — matrix operand order repaired and checked by reading, not compiled, not run on hardware
//
#include <hip/hip_runtime.h>
#include <math.h>

// ---------------------------------------------------------------- constants
constexpr int Bc   = 64;
constexpr int Sc   = 512;
constexpr int Hc   = 512;
constexpr int DFFc = 2048;
constexpr int HOPc = 6;
constexpr int BSc  = Bc * Sc;          // 32768 rows
constexpr float THRESH = 0.9f;

// FFN tiling
constexpr int MT        = 32;           // rows per workgroup (2 M-tiles of 16)
constexpr int NTW       = 4;            // n-tiles per wave
constexpr int XS_PITCH  = 520;          // halves per row of X tile   (512 + pad, 16B aligned)
constexpr int ACT_PITCH = 2056;         // halves per row of ACT tile (2048 + pad, 16B aligned)

constexpr size_t LDS_BYTES =
    (size_t)MT * ACT_PITCH * 2 +        // ACT tile (f16)  ~128.5 KB
    (size_t)MT * XS_PITCH * 2 +         // X tile   (f16)  ~32.5 KB
    (size_t)MT * 4;                     // uw per row (f32)

typedef __attribute__((ext_vector_type(16))) _Float16 v16h;
typedef __attribute__((ext_vector_type(8)))  float    v8f;

// -------------------------------------------------- WMMA fragment loaders
// A 16x32 f16 layout: lanes 0-15 -> M=lane,    K = {0..7, 16..23}
//                     lanes16-31 -> M=lane-16, K = {8..15,24..31}
__device__ __forceinline__ v16h load_a(const _Float16* base, int pitch,
                                       int r0, int k0, int lane) {
    const int l  = lane & 15;
    const int hi = lane >> 4;
    const _Float16* p = base + (size_t)(r0 + l) * pitch + k0 + hi * 8;
    v16h f;
#pragma unroll
    for (int i = 0; i < 8; ++i) { f[i] = p[i]; f[8 + i] = p[16 + i]; }
    return f;
}

// Pre-packed B fragment: tile = (ntile * K/32 + kblk); each lane owns 16
// contiguous halves (32 B) in the exact V_WMMA B operand order:
//   lanes 0-15 -> N=n0+lane,    K = k0+{0..7, 16..23}
//   lanes16-31 -> N=n0+lane-16, K = k0+{8..15,24..31}
__device__ __forceinline__ v16h load_bpk(const _Float16* __restrict__ wp,
                                         size_t tile, int lane) {
    return *(const v16h*)(wp + (tile * 32 + (size_t)lane) * 16);
}

__device__ __forceinline__ v8f wmma_f16(v16h a, v16h b, v8f c) {
    return __builtin_amdgcn_wmma_f32_16x16x32_f16(false, a, false, b,
                                                  (short)0, c, false, false);
}

// ---------------------------------------------------------------- kernels
// Pack fp32 weight W[K][N] (row-major) into per-lane B-fragment tiles (f16).
// One wave per 32x16 tile; 8 waves / block.
__global__ void pack_w(const float* __restrict__ w, _Float16* __restrict__ wp,
                       int N, int K) {
    const int lane = threadIdx.x & 31;
    const int wv   = threadIdx.x >> 5;
    const int tile = blockIdx.x * 8 + wv;           // ntile * (K/32) + kblk
    const int kpt  = K / 32;
    const int nt   = tile / kpt;
    const int kb   = tile - nt * kpt;
    const int n    = nt * 16 + (lane & 15);
    const int k0   = kb * 32 + (lane >> 4) * 8;
    _Float16* dst = wp + ((size_t)tile * 32 + lane) * 16;
#pragma unroll
    for (int i = 0; i < 16; ++i)
        dst[i] = (_Float16)w[(size_t)(k0 + (i < 8 ? i : i + 8)) * N + n];
}

__global__ void act_init(const float* __restrict__ state,
                         float* __restrict__ st, float* __restrict__ prev,
                         float* __restrict__ halting,
                         float* __restrict__ remainders,
                         float* __restrict__ n_updates) {
    const size_t i = (size_t)blockIdx.x * blockDim.x + threadIdx.x;
    if (i < (size_t)BSc * Hc) { st[i] = state[i]; prev[i] = 0.0f; }
    if (i < (size_t)BSc) { halting[i] = 0.0f; remainders[i] = 0.0f; n_updates[i] = 0.0f; }
}

// wave-per-row halting probability + ACT bookkeeping
__global__ void act_halt(const float* __restrict__ st,
                         const float* __restrict__ te,
                         const float* __restrict__ pe_t,
                         const float* __restrict__ w_p,
                         const float* __restrict__ b_p,
                         float* __restrict__ halting,
                         float* __restrict__ remainders,
                         float* __restrict__ n_updates,
                         float* __restrict__ uw) {
    const int lane = threadIdx.x & 31;
    const int wv   = threadIdx.x >> 5;
    const int row  = blockIdx.x * 8 + wv;
    const int s    = row & (Sc - 1);

    float acc = 0.0f;
#pragma unroll
    for (int j = 0; j < Hc / 32; ++j) {
        const int h = j * 32 + lane;
        const float x = st[(size_t)row * Hc + h] + te[(size_t)s * Hc + h] + pe_t[h];
        acc += x * w_p[h];
    }
#pragma unroll
    for (int off = 16; off > 0; off >>= 1) acc += __shfl_xor(acc, off, 32);

    if (lane == 0) {
        const float p = 1.0f / (1.0f + __expf(-(acc + b_p[0])));
        float hlt = halting[row], rem = remainders[row], nu = n_updates[row];
        float still = (hlt < 1.0f) ? 1.0f : 0.0f;
        const float cand = hlt + p * still;
        const float nh = ((cand > THRESH) ? 1.0f : 0.0f) * still;
        still = ((cand <= THRESH) ? 1.0f : 0.0f) * still;
        hlt += p * still;
        rem += nh * (1.0f - hlt);
        hlt += nh * rem;
        nu  += still + nh;
        halting[row]    = hlt;
        remainders[row] = rem;
        n_updates[row]  = nu;
        uw[row]         = p * still + nh * rem;
    }
}

// fused FFN: st_new = relu((st+te+pe) @ W1 + b1) @ W2 + b2 ; prev blend
// Weights arrive pre-packed (f16 B-fragment tiles, L2-resident).
// Per wave: 2 M-tiles x 4 N-tiles -> 8 WMMAs per (2 A + 4 B) fragment loads.
__global__ __launch_bounds__(256)
void act_ffn(float* __restrict__ st,
             const float* __restrict__ te,
             const float* __restrict__ pe_t,
             const _Float16* __restrict__ w1p, const float* __restrict__ b1,
             const _Float16* __restrict__ w2p, const float* __restrict__ b2,
             const float* __restrict__ uw_in,
             float* __restrict__ prev) {
    extern __shared__ char smem[];
    _Float16* ACTs = (_Float16*)smem;                       // [MT][ACT_PITCH]
    _Float16* Xs   = ACTs + (size_t)MT * ACT_PITCH;         // [MT][XS_PITCH]
    float*    uws  = (float*)(Xs + (size_t)MT * XS_PITCH);  // [MT]

    const int tid   = threadIdx.x;
    const int lane  = tid & 31;
    const int wv    = tid >> 5;                             // 0..7
    const int row0  = blockIdx.x * MT;

    // ---- stage X tile: (st + time_enc + pos_t) -> f16 in LDS
#pragma unroll
    for (int i = 0; i < (MT * Hc / 4) / 256; ++i) {         // 16 iters of float4
        const int idx4 = tid + i * 256;                      // 0..4095
        const int r  = idx4 >> 7;                            // /128
        const int c4 = idx4 & 127;
        const int row = row0 + r;
        const int s   = row & (Sc - 1);
        const float4 xv = *(const float4*)(st   + (size_t)row * Hc + c4 * 4);
        const float4 tv = *(const float4*)(te   + (size_t)s   * Hc + c4 * 4);
        const float4 pv = *(const float4*)(pe_t + c4 * 4);
        _Float16* dst = Xs + (size_t)r * XS_PITCH + c4 * 4;
        dst[0] = (_Float16)(xv.x + tv.x + pv.x);
        dst[1] = (_Float16)(xv.y + tv.y + pv.y);
        dst[2] = (_Float16)(xv.z + tv.z + pv.z);
        dst[3] = (_Float16)(xv.w + tv.w + pv.w);
    }
    if (tid < MT) uws[tid] = uw_in[row0 + tid];
    __syncthreads();

    // ---- phase 1: ACT = relu(X @ W1 + b1), kept in LDS as f16
    // 4 passes x (8 waves x 4 n-tiles x 16 cols) = 2048 DFF columns
    for (int np = 0; np < DFFc / (256 * NTW / 2); ++np) {    // 4 passes
        const int nt0 = np * (8 * NTW) + wv * NTW;           // first n-tile
        v8f c0[NTW], c1[NTW];
#pragma unroll
        for (int t = 0; t < NTW; ++t) {
            c0[t] = (v8f){0.f,0.f,0.f,0.f,0.f,0.f,0.f,0.f};
            c1[t] = (v8f){0.f,0.f,0.f,0.f,0.f,0.f,0.f,0.f};
        }
        for (int kb = 0; kb < Hc / 32; ++kb) {               // 16 K-steps
            const v16h fa0 = load_a(Xs, XS_PITCH, 0,  kb * 32, lane);
            const v16h fa1 = load_a(Xs, XS_PITCH, 16, kb * 32, lane);
#pragma unroll
            for (int t = 0; t < NTW; ++t) {
                const v16h fb = load_bpk(w1p, (size_t)(nt0 + t) * (Hc / 32) + kb, lane);
                c0[t] = wmma_f16(fa0, fb, c0[t]);
                c1[t] = wmma_f16(fa1, fb, c1[t]);
            }
        }
        // bias + relu -> f16 ACT tile
        const int mb = (lane >> 4) * 8;
#pragma unroll
        for (int t = 0; t < NTW; ++t) {
            const int cg = (nt0 + t) * 16 + (lane & 15);     // DFF column
            const float bias = b1[cg];
#pragma unroll
            for (int i = 0; i < 8; ++i) {
                ACTs[(size_t)(mb + i)      * ACT_PITCH + cg] =
                    (_Float16)fmaxf(c0[t][i] + bias, 0.0f);
                ACTs[(size_t)(16 + mb + i) * ACT_PITCH + cg] =
                    (_Float16)fmaxf(c1[t][i] + bias, 0.0f);
            }
        }
    }
    __syncthreads();

    // ---- phase 2: st_new = ACT @ W2 + b2 ; prev blend ; write st
    // single pass: 8 waves x 4 n-tiles x 16 = 512 H columns
    {
        const int nt0 = wv * NTW;
        v8f c0[NTW], c1[NTW];
#pragma unroll
        for (int t = 0; t < NTW; ++t) {
            c0[t] = (v8f){0.f,0.f,0.f,0.f,0.f,0.f,0.f,0.f};
            c1[t] = (v8f){0.f,0.f,0.f,0.f,0.f,0.f,0.f,0.f};
        }
        for (int kb = 0; kb < DFFc / 32; ++kb) {             // 64 K-steps
            const v16h fa0 = load_a(ACTs, ACT_PITCH, 0,  kb * 32, lane);
            const v16h fa1 = load_a(ACTs, ACT_PITCH, 16, kb * 32, lane);
#pragma unroll
            for (int t = 0; t < NTW; ++t) {
                const v16h fb = load_bpk(w2p, (size_t)(nt0 + t) * (DFFc / 32) + kb, lane);
                c0[t] = wmma_f16(fa0, fb, c0[t]);
                c1[t] = wmma_f16(fa1, fb, c1[t]);
            }
        }
        const int mb = (lane >> 4) * 8;
#pragma unroll
        for (int t = 0; t < NTW; ++t) {
            const int cg = (nt0 + t) * 16 + (lane & 15);     // H column
            const float bias = b2[cg];
#pragma unroll
            for (int i = 0; i < 8; ++i) {
                {   // rows 0..15 tile
                    const int r = mb + i;
                    const float v = c0[t][i] + bias;
                    const size_t gi = (size_t)(row0 + r) * Hc + cg;
                    const float u = uws[r];
                    st[gi]   = v;
                    prev[gi] = v * u + prev[gi] * (1.0f - u);
                }
                {   // rows 16..31 tile
                    const int r = 16 + mb + i;
                    const float v = c1[t][i] + bias;
                    const size_t gi = (size_t)(row0 + r) * Hc + cg;
                    const float u = uws[r];
                    st[gi]   = v;
                    prev[gi] = v * u + prev[gi] * (1.0f - u);
                }
            }
        }
    }
}

__global__ void act_final(const float* __restrict__ remainders,
                          const float* __restrict__ n_updates,
                          float* __restrict__ out_tail) {
    const int i = blockIdx.x * blockDim.x + threadIdx.x;
    if (i < BSc) {
        out_tail[i]       = remainders[i];
        out_tail[BSc + i] = n_updates[i];
    }
}

// ---------------------------------------------------------------- launcher
extern "C" void kernel_launch(void* const* d_in, const int* in_sizes, int n_in,
                              void* d_out, int out_size, void* d_ws, size_t ws_size,
                              hipStream_t stream) {
    const float* state = (const float*)d_in[0];
    // d_in[1] ("inputs") only defines shapes in the reference — unused.
    const float* te  = (const float*)d_in[2];
    const float* pe  = (const float*)d_in[3];
    const float* w_p = (const float*)d_in[4];
    const float* b_p = (const float*)d_in[5];
    const float* w1  = (const float*)d_in[6];
    const float* b1  = (const float*)d_in[7];
    const float* w2  = (const float*)d_in[8];
    const float* b2  = (const float*)d_in[9];

    float* out  = (float*)d_out;
    float* prev = out;                                   // [BS*H]
    float* st         = (float*)d_ws;                    // [BS*H]
    float* halting    = st + (size_t)BSc * Hc;           // [BS]
    float* remainders = halting + BSc;                   // [BS]
    float* n_updates  = remainders + BSc;                // [BS]
    float* uw         = n_updates + BSc;                 // [BS]
    _Float16* w1p = (_Float16*)(uw + BSc);               // [H * DFF] packed
    _Float16* w2p = w1p + (size_t)Hc * DFFc;             // [DFF * H] packed

    // one-time per launch: pack weights into B-fragment layout (f16)
    pack_w<<<(DFFc / 16) * (Hc / 32) / 8, 256, 0, stream>>>(w1, w1p, DFFc, Hc);
    pack_w<<<(Hc / 16) * (DFFc / 32) / 8, 256, 0, stream>>>(w2, w2p, Hc, DFFc);

    act_init<<<(BSc * Hc) / 256, 256, 0, stream>>>(state, st, prev,
                                                   halting, remainders, n_updates);
    for (int t = 0; t < HOPc; ++t) {
        const float* pe_t = pe + (size_t)t * Hc;
        act_halt<<<BSc / 8, 256, 0, stream>>>(st, te, pe_t, w_p, b_p,
                                              halting, remainders, n_updates, uw);
        act_ffn<<<BSc / MT, 256, LDS_BYTES, stream>>>(st, te, pe_t,
                                                      w1p, b1, w2p, b2, uw, prev);
    }
    act_final<<<BSc / 256, 256, 0, stream>>>(remainders, n_updates,
                                             out + (size_t)BSc * Hc);
}
